// TextEncoder_21311627722945
// MI455X (gfx1250) — matrix operand order repaired and checked by reading, not compiled, not run on hardware
//
#include <hip/hip_runtime.h>
#include <cstdint>
#include <cstddef>

// Problem constants (from reference)
#define EMBED   512
#define UNITS   512
#define GATES   2048      // 4*UNITS
#define NSEQ    256       // B*S
#define TLEN    128
#define KDIM    1024      // EMBED + UNITS  (fused [W;U] K dimension)
#define HIDDEN  1024
#define LDS_STRIDE 1032   // KDIM + 8 halfs pad -> 2064B row stride (bank-conflict free)

typedef __bf16 v16bf __attribute__((ext_vector_type(16)));
typedef float  v8f   __attribute__((ext_vector_type(8)));

__device__ __forceinline__ unsigned short f2bf(float f) {
  __bf16 h = (__bf16)f;                       // RNE convert
  return __builtin_bit_cast(unsigned short, h);
}
__device__ __forceinline__ float sigmoidf_(float x) {
  return 1.0f / (1.0f + __expf(-x));
}

// ---------------------------------------------------------------------------
// Kernel 1: combined, transposed bf16 weights per direction:
//   Bt[dir][col][k] = bf16( k<512 ? W[k][col] : U[k-512][col] )
// Transposed layout -> each WMMA B-fragment is one contiguous 32B load/lane.
// ---------------------------------------------------------------------------
__global__ __launch_bounds__(256) void prep_weights_kernel(
    const float* __restrict__ W_f, const float* __restrict__ U_f,
    const float* __restrict__ W_b, const float* __restrict__ U_b,
    unsigned short* __restrict__ Bt) {
  size_t idx = (size_t)blockIdx.x * 256 + threadIdx.x;   // 2*2048*1024 total
  int dir = (int)(idx >> 21);
  int rem = (int)(idx & ((1u << 21) - 1));
  int col = rem >> 10;
  int k   = rem & 1023;
  const float* W = dir ? W_b : W_f;
  const float* U = dir ? U_b : U_f;
  float v = (k < EMBED) ? W[(size_t)k * GATES + col]
                        : U[(size_t)(k - EMBED) * GATES + col];
  Bt[idx] = f2bf(v);
}

// ---------------------------------------------------------------------------
// Kernel 2: embedding gather + bf16 convert.  One block per (n,t) row.
// ---------------------------------------------------------------------------
__global__ __launch_bounds__(256) void gather_embed_kernel(
    const int* __restrict__ x, const float* __restrict__ table,
    unsigned short* __restrict__ embBF) {
  int nt = blockIdx.x;                       // 0 .. 32767
  int tok = x[nt];
  const float* src = table + (size_t)tok * EMBED;
  unsigned short* dst = embBF + (size_t)nt * EMBED;
  for (int k = threadIdx.x; k < EMBED; k += 256) dst[k] = f2bf(src[k]);
}

// ---------------------------------------------------------------------------
// Kernel 3: pack mask (token != 0) into 4 u32 words per sequence.
// ---------------------------------------------------------------------------
__global__ __launch_bounds__(256) void make_mask_kernel(
    const int* __restrict__ x, unsigned* __restrict__ maskbits) {
  int id = blockIdx.x * 256 + threadIdx.x;   // 1024 total
  if (id >= NSEQ * 4) return;
  int n = id >> 2, w = id & 3;
  unsigned bits = 0;
  for (int j = 0; j < 32; j++)
    bits |= (x[n * TLEN + w * 32 + j] != 0 ? 1u : 0u) << j;
  maskbits[id] = bits;
}

// ---------------------------------------------------------------------------
// Kernel 4: persistent bidirectional LSTM recurrence.
// Grid = 32 WGs: blockIdx.x = dir + 2*row_group (16 sequences per WG).
// 512 threads = 16 waves; wave owns 32 units (2 unit-tiles x 4 gates = 8
// accumulators).  Per step:
//   z = [emb_t | h] (16x1024 bf16, LDS) @ Bt (1024x2048 bf16, L2-resident)
// f32-accumulated via v_wmma_f32_16x16x32_bf16 with B fragments software-
// pipelined THREE tiles ahead (4-slot ring), then fused gate math with
// Keras masking (state held where token==0).
// ---------------------------------------------------------------------------
__global__ __launch_bounds__(512, 1) void bilstm_kernel(
    const unsigned short* __restrict__ embBF,   // [NSEQ][TLEN][EMBED] bf16
    const unsigned short* __restrict__ Bt,      // [2][GATES][KDIM]    bf16
    const unsigned* __restrict__ maskbits,      // [NSEQ][4]
    const float* __restrict__ b_f, const float* __restrict__ b_b,
    float* __restrict__ out) {                  // [NSEQ][HIDDEN]
  __shared__ unsigned short lds[16 * LDS_STRIDE];  // A rows: [emb(512)|h(512)|pad]

  const int dir  = blockIdx.x & 1;
  const int rg   = blockIdx.x >> 1;
  const int n0   = rg * 16;
  const int tid  = threadIdx.x;
  const int wave = tid >> 5;       // 0..15
  const int lane = tid & 31;
  const int hi   = lane >> 4;      // K-half select (A/B layouts)
  const int l16  = lane & 15;      // A-row / B-col / D-col index
  const int ub   = wave * 32;      // this wave's first unit column

  const unsigned short* B = Bt + (size_t)dir * GATES * KDIM;
  const float* bias = dir ? b_b : b_f;

  // Per-tile B base pointers (col-major over K): tile -> (ut,g).
  const unsigned short* bbase[8];
#pragma unroll
  for (int tile = 0; tile < 8; tile++) {
    const int ut = tile >> 2, g = tile & 3;
    int col = g * UNITS + ub + ut * 16 + l16;
    bbase[tile] = B + (size_t)col * KDIM + hi * 16;
  }

  // Bias per (unit-tile, gate): constant C-init each step (z = xW + b + hU).
  float bval[2][4];
#pragma unroll
  for (int ut = 0; ut < 2; ut++)
#pragma unroll
    for (int g = 0; g < 4; g++)
      bval[ut][g] = bias[g * UNITS + ub + ut * 16 + l16];

  // Zero the h half of the A buffer.
  for (int i = tid; i < 16 * UNITS; i += 512) {
    int r = i >> 9, k = i & 511;
    lds[r * LDS_STRIDE + EMBED + k] = 0;
  }

  v8f c[2] = {}, h[2] = {};

  for (int s = 0; s < TLEN; s++) {
    const int t = dir ? (TLEN - 1 - s) : s;

    // Stage emb_t (16 rows x 512 halfs = 16KB): 32B per thread.
    {
      int r = tid >> 5, ch = tid & 31;   // 16 rows x 32 chunks of 16 halfs
      const uint4* src = (const uint4*)(embBF +
          ((size_t)(n0 + r) * TLEN + t) * EMBED + ch * 16);
      uint4* dst = (uint4*)(lds + r * LDS_STRIDE + ch * 16);
      dst[0] = src[0]; dst[1] = src[1];
    }
    __syncthreads();   // emb_t staged AND previous-step h writes visible

    // Accumulators: 2 unit tiles x 4 gates, C-init = bias broadcast per col.
    v8f acc[2][4];
#pragma unroll
    for (int ut = 0; ut < 2; ut++)
#pragma unroll
      for (int g = 0; g < 4; g++) {
        float bv = bval[ut][g];
#pragma unroll
        for (int e = 0; e < 8; e++) acc[ut][g][e] = bv;
      }

    union BF { uint4 u[2]; v16bf v; };

#pragma unroll 1
    for (int kt = 0; kt < 32; kt++) {          // K = 1024 = 32 tiles of 32
      const int kb = kt * 32;                  // k offset in halfs
      // A fragment (16x32 bf16): lanes 0-15 K{0-15}, lanes 16-31 K{16-31}
      v16bf a;
      {
        union { uint4 u[2]; v16bf v; } cv;
        cv.u[0] = *(const uint4*)(lds + l16 * LDS_STRIDE + kb + hi * 16);
        cv.u[1] = *(const uint4*)(lds + l16 * LDS_STRIDE + kb + hi * 16 + 8);
        a = cv.v;
      }
      // Prefetch next k-tile of this wave's B stream (global_prefetch_b8).
      __builtin_prefetch(B + (size_t)(ub + l16) * KDIM + ((kt + 1) & 31) * 32, 0, 1);

      // Software pipeline, depth 3: B fragments for tiles i+1..i+3 are in
      // flight while WMMA i runs (4-slot ring; WAR on the ring bounds hoisting).
      BF bfr[4];
#pragma unroll
      for (int p = 0; p < 3; p++) {
        const uint4* bp = (const uint4*)(bbase[p] + kb);
        bfr[p].u[0] = bp[0];
        bfr[p].u[1] = bp[1];
      }
#pragma unroll
      for (int tile = 0; tile < 8; tile++) {
        const int ut = tile >> 2, g = tile & 3;
        if (tile + 3 < 8) {
          const uint4* bp = (const uint4*)(bbase[tile + 3] + kb);
          bfr[(tile + 3) & 3].u[0] = bp[0];
          bfr[(tile + 3) & 3].u[1] = bp[1];
        }
        acc[ut][g] = __builtin_amdgcn_wmma_f32_16x16x32_bf16(
            false, a, false, bfr[tile & 3].v, (short)0, acc[ut][g], false, false);
        // Keep pipeline shape stable; unbounded load batching caused the
        // round-1 scratch spills.
        __builtin_amdgcn_sched_barrier(0);
      }
    }
    __syncthreads();   // all A reads done before h / next emb overwrite

    // Gate math + masked state update.  D layout: row = e + 8*hi, col = l16.
#pragma unroll
    for (int ut = 0; ut < 2; ut++) {
#pragma unroll
      for (int e = 0; e < 8; e++) {
        int row = e + 8 * hi;
        int n   = n0 + row;
        unsigned mword = maskbits[n * 4 + (t >> 5)];
        bool m = (mword >> (t & 31)) & 1u;

        float iv = sigmoidf_(acc[ut][0][e]);
        float fv = sigmoidf_(acc[ut][1][e]);
        float gv = tanhf(acc[ut][2][e]);
        float ov = sigmoidf_(acc[ut][3][e]);
        float cn = fv * c[ut][e] + iv * gv;
        float hn = ov * tanhf(cn);
        float cw = m ? cn : c[ut][e];
        float hw = m ? hn : h[ut][e];
        c[ut][e] = cw;
        h[ut][e] = hw;
        lds[row * LDS_STRIDE + EMBED + ub + ut * 16 + l16] = f2bf(hw);
      }
    }
  }

  // Final hidden state -> out[n][dir*512 + unit] (fwd | bwd concat).
#pragma unroll
  for (int ut = 0; ut < 2; ut++)
#pragma unroll
    for (int e = 0; e < 8; e++) {
      int row = e + 8 * hi;
      out[(size_t)(n0 + row) * HIDDEN + dir * UNITS + ub + ut * 16 + l16] = h[ut][e];
    }
}

// ---------------------------------------------------------------------------
// Launch.  Inputs (setup_inputs order):
//   0:x(int32) 1:emb_table(f32) 2:W_f 3:U_f 4:b_f 5:W_b 6:U_b 7:b_b
// Workspace layout: embBF (32MB) | Bt (8MB) | maskbits (4KB)  ~= 42MB
// ---------------------------------------------------------------------------
extern "C" void kernel_launch(void* const* d_in, const int* in_sizes, int n_in,
                              void* d_out, int out_size, void* d_ws, size_t ws_size,
                              hipStream_t stream) {
  const int*   x         = (const int*)d_in[0];
  const float* emb_table = (const float*)d_in[1];
  const float* W_f       = (const float*)d_in[2];
  const float* U_f       = (const float*)d_in[3];
  const float* b_f       = (const float*)d_in[4];
  const float* W_b       = (const float*)d_in[5];
  const float* U_b       = (const float*)d_in[6];
  const float* b_b       = (const float*)d_in[7];
  float*       out       = (float*)d_out;

  char* w = (char*)d_ws;
  unsigned short* embBF = (unsigned short*)w;                  // 32768*512*2
  w += (size_t)NSEQ * TLEN * EMBED * 2;
  unsigned short* Bt = (unsigned short*)w;                     // 2*2048*1024*2
  w += (size_t)2 * GATES * KDIM * 2;
  unsigned* maskbits = (unsigned*)w;                           // 256*4*4

  prep_weights_kernel<<<(2 * GATES * KDIM) / 256, 256, 0, stream>>>(
      W_f, U_f, W_b, U_b, Bt);
  gather_embed_kernel<<<NSEQ * TLEN, 256, 0, stream>>>(x, emb_table, embBF);
  make_mask_kernel<<<4, 256, 0, stream>>>(x, maskbits);
  bilstm_kernel<<<32, 512, 0, stream>>>(embBF, Bt, maskbits, b_f, b_b, out);
}
